// DecoderLayerTorch_55456617726533
// MI455X (gfx1250) — compile-verified, matrix operand order checked by reading, not run on hardware
//
#include <hip/hip_runtime.h>
#include <hip/hip_bf16.h>

// ---------------- constants from the reference ----------------
#define BB   16
#define TT   64
#define EE   2048
#define HH   32
#define IMGI 256
#define GG   4096
#define DD   64
// ---------------------------------------------------------------

typedef __attribute__((ext_vector_type(16))) __bf16       v16bf;
typedef __attribute__((ext_vector_type(8)))  float        v8f;
typedef __attribute__((ext_vector_type(8)))  unsigned int v8u;

__device__ __forceinline__ unsigned int pack_bf16_pair(float lo, float hi) {
  unsigned int ul = __builtin_bit_cast(unsigned int, lo);
  unsigned int uh = __builtin_bit_cast(unsigned int, hi);
  ul += 0x7fffu + ((ul >> 16) & 1u);          // RNE
  uh += 0x7fffu + ((uh >> 16) & 1u);
  return (ul >> 16) | (uh & 0xffff0000u);
}

__device__ __forceinline__ float gelu_exact(float x) {
  return 0.5f * x * (1.0f + erff(x * 0.70710678118654752f));
}

// K index for fragment dword j, lane-half h (CDNA5 16-bit A/B layout)
__device__ __forceinline__ int frag_k(int j, int h) {
  return ((j >> 2) << 4) + (h << 3) + ((j & 3) << 1);
}

// =====================================================================
// WMMA GEMM: C[M x N] = act(alpha * A[M x K] @ W[K x N])
// Block = 256 threads = 8 waves; tile = 16 (M) x 128 (N); K step = 32.
// LDS tiles are stored in *fragment order* so each lane builds its
// 16x32 fragment with two ds_load_b128. Staging uses loop-invariant
// element OFFSETS from the global base pointers (keeps address-space
// inference -> global_load, not flat_load) bumped by constant strides.
// blockIdx.z selects a K chunk (split-K); partials go to C[z*M*N+..]
// and a deterministic reduce applies alpha/activation.
// =====================================================================
#define ROWSTRIDE 20   // dwords per A-row / B-col slot (16 used + 4 pad)

__global__ __launch_bounds__(256)
void gemm_wmma_bf16(const float* __restrict__ A, const float* __restrict__ W,
                    float* __restrict__ C, int M, int N, int K,
                    float alpha, int act, int kchunk)
{
  __shared__ unsigned int As2[16  * ROWSTRIDE];   // 1.25 KB
  __shared__ unsigned int Bs2[128 * ROWSTRIDE];   // 10  KB

  const int tid  = threadIdx.x;
  const int wave = tid >> 5;
  const int lane = tid & 31;
  const int m0   = blockIdx.y * 16;
  const int n0   = blockIdx.x * 128;

  const int kstart = blockIdx.z * kchunk;
  const int kend   = min(K, kstart + kchunk);

  v8f acc = (v8f)(0.0f);

  const int mrow = lane & 15;            // M index inside 16x16 tile
  const int half = lane >> 4;            // K-half selector per layout
  const int col  = (wave << 4) + mrow;   // N column inside 128-wide tile

  // ---- loop-invariant staging offsets (element units) ----
  // A: 256 threads -> (m, h, j), one packed dword each per k-step.
  const int a_m = tid >> 4;
  const int a_h = (tid >> 3) & 1;
  const int a_j = tid & 7;
  size_t aoff = (size_t)(m0 + a_m) * K + kstart + frag_k(a_j, a_h);
  const int a_dst = a_m * ROWSTRIDE + a_h * 8 + a_j;

  // B: 8 slots per thread, offsets bumped by 32*N elements per k-step.
  size_t boff[8];
  int bdst[8];
  #pragma unroll
  for (int i = 0; i < 8; ++i) {
    int e   = tid + i * 256;
    int c   = e & 127;
    int sub = e >> 7;
    int h   = sub >> 3;
    int j   = sub & 7;
    boff[i] = (size_t)(kstart + frag_k(j, h)) * N + n0 + c;
    bdst[i] = c * ROWSTRIDE + h * 8 + j;
  }
  const size_t wstep = (size_t)32 * N;

  for (int k0 = kstart; k0 < kend; k0 += 32) {
    // --- stage A tile (16x32): 1 packed dword/thread ---
    As2[a_dst] = pack_bf16_pair(A[aoff], A[aoff + 1]);
    aoff += 32;
    // --- stage B tile (32x128): 8 packed dwords/thread ---
    #pragma unroll
    for (int i = 0; i < 8; ++i) {
      Bs2[bdst[i]] = pack_bf16_pair(W[boff[i]], W[boff[i] + N]);
    }
    // pull one cacheline of the next tile toward GL2 per thread
    if (k0 + 32 < kend) __builtin_prefetch(W + boff[0] + wstep, 0, 1);
    #pragma unroll
    for (int i = 0; i < 8; ++i) boff[i] += wstep;
    __syncthreads();

    // --- fragment build: two b128 loads each for A and B ---
    const unsigned int* ap = &As2[mrow * ROWSTRIDE + half * 8];
    const unsigned int* bp = &Bs2[col  * ROWSTRIDE + half * 8];
    uint4 a0 = *reinterpret_cast<const uint4*>(ap);
    uint4 a1 = *reinterpret_cast<const uint4*>(ap + 4);
    uint4 b0 = *reinterpret_cast<const uint4*>(bp);
    uint4 b1 = *reinterpret_cast<const uint4*>(bp + 4);
    v8u au = {a0.x, a0.y, a0.z, a0.w, a1.x, a1.y, a1.z, a1.w};
    v8u bu = {b0.x, b0.y, b0.z, b0.w, b1.x, b1.y, b1.z, b1.w};
    v16bf a = __builtin_bit_cast(v16bf, au);
    v16bf b = __builtin_bit_cast(v16bf, bu);

    acc = __builtin_amdgcn_wmma_f32_16x16x32_bf16(
        /*neg_a=*/false, a, /*neg_b=*/false, b,
        /*c_mod=*/(short)0, acc, /*reuse_a=*/false, /*reuse_b=*/false);
    __syncthreads();
  }

  // --- epilogue: C layout m = r + 8*(lane>>4), n = lane&15 ---
  const int nOut = n0 + col;
  const size_t MN = (size_t)M * N;
  #pragma unroll
  for (int r = 0; r < 8; ++r) {
    int m = m0 + r + (half << 3);
    if (gridDim.z == 1) {
      float v = acc[r] * alpha;
      if (act == 1) v = gelu_exact(v);
      C[(size_t)m * N + nOut] = v;
    } else {
      C[(size_t)blockIdx.z * MN + (size_t)m * N + nOut] = acc[r];
    }
  }
}

// Deterministic split-K reduction + alpha/activation epilogue.
__global__ void reduce_parts_kernel(const float* __restrict__ part,
                                    float* __restrict__ out,
                                    int n, int nz, float alpha, int act)
{
  int i = blockIdx.x * 256 + threadIdx.x;
  if (i >= n) return;
  float s = 0.0f;
  for (int z = 0; z < nz; ++z) s += part[(size_t)z * n + i];
  s *= alpha;
  if (act == 1) s = gelu_exact(s);
  out[i] = s;
}

// =====================================================================
// LayerNorm over last dim Wd (one block per row); optional residual add.
// =====================================================================
__global__ __launch_bounds__(256)
void layernorm_kernel(const float* __restrict__ x, const float* __restrict__ g,
                      const float* __restrict__ bta, const float* __restrict__ res,
                      float* __restrict__ out, int Wd)
{
  const int row = blockIdx.x;
  const int tid = threadIdx.x;
  const float* xr = x + (size_t)row * Wd;
  float s1 = 0.0f, s2 = 0.0f;
  for (int c = tid; c < Wd; c += 256) { float v = xr[c]; s1 += v; s2 += v * v; }
  __shared__ float r1[256], r2[256];
  r1[tid] = s1; r2[tid] = s2;
  __syncthreads();
  for (int off = 128; off > 0; off >>= 1) {
    if (tid < off) { r1[tid] += r1[tid + off]; r2[tid] += r2[tid + off]; }
    __syncthreads();
  }
  const float mean = r1[0] / (float)Wd;
  const float var  = r2[0] / (float)Wd - mean * mean;
  const float rstd = rsqrtf(var + 1e-5f);
  float* orow = out + (size_t)row * Wd;
  const float* rr = res ? res + (size_t)row * Wd : nullptr;
  for (int c = tid; c < Wd; c += 256) {
    float v = (xr[c] - mean) * rstd * g[c] + bta[c];
    if (rr) v += rr[c];
    orow[c] = v;
  }
}

// Scatter k_new/v_new into the token_index row of the output KV cache.
__global__ void kv_scatter_kernel(const float* __restrict__ knew,
                                  const float* __restrict__ vnew,
                                  const int* __restrict__ tokp,
                                  float* __restrict__ kvout)
{
  int i = blockIdx.x * 256 + threadIdx.x;
  if (i >= BB * EE) return;
  int b = i >> 11, j = i & (EE - 1);
  int tok = *tokp;
  kvout[((size_t)(b * IMGI) + tok) * EE + j]        = knew[i];
  kvout[((size_t)((BB + b) * IMGI) + tok) * EE + j] = vnew[i];
}

// Self-attention: one block per (b,h); 256 threads = 1 thread per key.
__global__ __launch_bounds__(256)
void self_attn_kernel(const float* __restrict__ q, const float* __restrict__ kvout,
                      const int* __restrict__ tokp, float* __restrict__ attn)
{
  const int b = blockIdx.x >> 5;
  const int h = blockIdx.x & 31;
  const int tid = threadIdx.x;
  const int tok = *tokp;
  __shared__ float sq[DD], sw[IMGI], red[256];
  if (tid < DD) sq[tid] = q[(size_t)b * EE + h * DD + tid];
  __syncthreads();
  const float* keyrow = kvout + ((size_t)b * IMGI + tid) * EE + h * DD;
  float s = 0.0f;
  #pragma unroll 8
  for (int d = 0; d < DD; ++d) s += sq[d] * keyrow[d];
  float sc = (tid <= tok) ? s : -1e30f;
  red[tid] = sc; __syncthreads();
  for (int off = 128; off > 0; off >>= 1) {
    if (tid < off) red[tid] = fmaxf(red[tid], red[tid + off]);
    __syncthreads();
  }
  float mx = red[0]; __syncthreads();
  float e = __expf(sc - mx);
  red[tid] = e; __syncthreads();
  for (int off = 128; off > 0; off >>= 1) {
    if (tid < off) red[tid] += red[tid + off];
    __syncthreads();
  }
  sw[tid] = e / red[0];
  __syncthreads();
  if (tid < DD) {
    const float* vbase = kvout + ((size_t)(BB + b) * IMGI) * EE + h * DD + tid;
    float acc = 0.0f;
    for (int k = 0; k < IMGI; ++k) acc += sw[k] * vbase[(size_t)k * EE];
    attn[(size_t)b * EE + h * DD + tid] = acc;
  }
}

// Cross-attention: one block per (b,h); 64 threads = 1 thread per key.
__global__ __launch_bounds__(64)
void cross_attn_kernel(const float* __restrict__ cq, const float* __restrict__ ck,
                       const float* __restrict__ cv,
                       const unsigned char* __restrict__ mask,
                       float* __restrict__ attn)
{
  const int b = blockIdx.x >> 5;
  const int h = blockIdx.x & 31;
  const int tid = threadIdx.x;
  __shared__ float sq[DD], sw[TT], red[TT];
  sq[tid] = cq[(size_t)b * EE + h * DD + tid];
  __syncthreads();
  const float* keyrow = ck + ((size_t)b * TT + tid) * EE + h * DD;
  float s = 0.0f;
  #pragma unroll 8
  for (int d = 0; d < DD; ++d) s += sq[d] * keyrow[d];
  float sc = mask[b * TT + tid] ? s : -1e30f;
  red[tid] = sc; __syncthreads();
  for (int off = 32; off > 0; off >>= 1) {
    if (tid < off) red[tid] = fmaxf(red[tid], red[tid + off]);
    __syncthreads();
  }
  float mx = red[0]; __syncthreads();
  float e = __expf(sc - mx);
  red[tid] = e; __syncthreads();
  for (int off = 32; off > 0; off >>= 1) {
    if (tid < off) red[tid] += red[tid + off];
    __syncthreads();
  }
  sw[tid] = e / red[0];
  __syncthreads();
  const float* vbase = cv + ((size_t)b * TT) * EE + h * DD + tid;
  float acc = 0.0f;
  for (int k = 0; k < TT; ++k) acc += sw[k] * vbase[(size_t)k * EE];
  attn[(size_t)b * EE + h * DD + tid] = acc;
}

__global__ void mul_kernel(const float* __restrict__ a, const float* __restrict__ b,
                           float* __restrict__ o, int n)
{
  int i = blockIdx.x * 256 + threadIdx.x;
  if (i < n) o[i] = a[i] * b[i];
}

__global__ void add_kernel(const float* __restrict__ a, const float* __restrict__ b,
                           float* __restrict__ o, int n)
{
  int i = blockIdx.x * 256 + threadIdx.x;
  if (i < n) o[i] = a[i] + b[i];
}

// ---------------- host side ----------------
static void launch_gemm(const float* A, const float* W, float* C, float* part,
                        int M, int N, int K, float alpha, int act, int nsplit,
                        hipStream_t stream)
{
  dim3 block(256);
  if (nsplit <= 1) {
    dim3 grid(N / 128, M / 16, 1);
    gemm_wmma_bf16<<<grid, block, 0, stream>>>(A, W, C, M, N, K, alpha, act, K);
  } else {
    int kchunk = K / nsplit;  // multiple of 32 for our shapes
    dim3 grid(N / 128, M / 16, nsplit);
    gemm_wmma_bf16<<<grid, block, 0, stream>>>(A, W, part, M, N, K, alpha, act, kchunk);
    int n = M * N;
    reduce_parts_kernel<<<(n + 255) / 256, 256, 0, stream>>>(part, C, n, nsplit, alpha, act);
  }
}

extern "C" void kernel_launch(void* const* d_in, const int* in_sizes, int n_in,
                              void* d_out, int out_size, void* d_ws, size_t ws_size,
                              hipStream_t stream) {
  const float* decoder = (const float*)d_in[0];
  const float* encoder = (const float*)d_in[1];
  const float* kv_in   = (const float*)d_in[2];
  const unsigned char* mask = (const unsigned char*)d_in[3];
  const int*   tokp    = (const int*)d_in[4];

  const float* ln_pre_sa_g = (const float*)d_in[5];
  const float* ln_pre_sa_b = (const float*)d_in[6];
  const float* ln_sa_g     = (const float*)d_in[7];
  const float* ln_sa_b     = (const float*)d_in[8];
  const float* sa_wk       = (const float*)d_in[9];
  const float* sa_wv       = (const float*)d_in[10];
  const float* sa_wq       = (const float*)d_in[11];
  const float* sa_wo       = (const float*)d_in[12];
  const float* ln_pre_ca_g = (const float*)d_in[13];
  const float* ln_pre_ca_b = (const float*)d_in[14];
  const float* ln_ca_g     = (const float*)d_in[15];
  const float* ln_ca_b     = (const float*)d_in[16];
  const float* ca_wk       = (const float*)d_in[17];
  const float* ca_wv       = (const float*)d_in[18];
  const float* ca_wq       = (const float*)d_in[19];
  const float* ca_wo       = (const float*)d_in[20];
  const float* glu_ln0_g   = (const float*)d_in[21];
  const float* glu_ln0_b   = (const float*)d_in[22];
  const float* glu_ln1_g   = (const float*)d_in[23];
  const float* glu_ln1_b   = (const float*)d_in[24];
  const float* glu_fc0     = (const float*)d_in[25];
  const float* glu_fc1     = (const float*)d_in[26];
  const float* glu_fc2     = (const float*)d_in[27];

  float* out_dec = (float*)d_out;                    // [16, 1, 2048]
  float* kv_out  = (float*)d_out + (size_t)BB * EE;  // [32, 256, 32, 64]

  float* ws = (float*)d_ws;
  const size_t BE = (size_t)BB * EE;      // 32768
  const size_t BG = (size_t)BB * GG;      // 65536
  float* x0   = ws;                 // BE each below
  float* q    = ws + BE;
  float* knew = ws + 2 * BE;
  float* vnew = ws + 3 * BE;
  float* attn = ws + 4 * BE;
  float* t0   = ws + 5 * BE;
  float* ds1  = ws + 6 * BE;
  float* x1   = ws + 7 * BE;
  float* cq   = ws + 8 * BE;
  float* at2  = ws + 9 * BE;
  float* t1   = ws + 10 * BE;
  float* ds2  = ws + 11 * BE;
  float* z    = ws + 12 * BE;
  float* t2   = ws + 13 * BE;
  float* wg   = ws + 14 * BE;       // BG each below
  float* vg   = wg + BG;
  float* hpre = vg + BG;
  float* hbuf = hpre + BG;
  float* part = hbuf + BG;          // up to 16 partial slabs
  float* ck   = part + 16 * BG;     // B*T*E
  float* cv   = ck + (size_t)BB * TT * EE;

  const float inv_sqrt_d = 0.125f;  // 1/sqrt(64)

  // KV cache passthrough (64 MB D2D) — legal under graph capture.
  hipMemcpyAsync(kv_out, kv_in, (size_t)2 * BB * IMGI * HH * DD * sizeof(float),
                 hipMemcpyDeviceToDevice, stream);

  // ---- self-attention block ----
  layernorm_kernel<<<BB, 256, 0, stream>>>(decoder, ln_pre_sa_g, ln_pre_sa_b, nullptr, x0, EE);
  launch_gemm(x0, sa_wk, knew, part, BB, EE, EE, 1.0f, 0, 16, stream);
  launch_gemm(x0, sa_wv, vnew, part, BB, EE, EE, 1.0f, 0, 16, stream);
  launch_gemm(x0, sa_wq, q,    part, BB, EE, EE, inv_sqrt_d, 0, 16, stream);
  kv_scatter_kernel<<<(BB * EE + 255) / 256, 256, 0, stream>>>(knew, vnew, tokp, kv_out);
  self_attn_kernel<<<BB * HH, 256, 0, stream>>>(q, kv_out, tokp, attn);
  launch_gemm(attn, sa_wo, t0, part, BB, EE, EE, 1.0f, 0, 16, stream);
  layernorm_kernel<<<BB, 256, 0, stream>>>(t0, ln_sa_g, ln_sa_b, decoder, ds1, EE);

  // ---- cross-attention block ----
  layernorm_kernel<<<BB, 256, 0, stream>>>(ds1, ln_pre_ca_g, ln_pre_ca_b, nullptr, x1, EE);
  launch_gemm(encoder, ca_wk, ck, part, BB * TT, EE, EE, 1.0f, 0, 1, stream);
  launch_gemm(encoder, ca_wv, cv, part, BB * TT, EE, EE, 1.0f, 0, 1, stream);
  launch_gemm(x1, ca_wq, cq, part, BB, EE, EE, inv_sqrt_d, 0, 16, stream);
  cross_attn_kernel<<<BB * HH, 64, 0, stream>>>(cq, ck, cv, mask, at2);
  launch_gemm(at2, ca_wo, t1, part, BB, EE, EE, 1.0f, 0, 16, stream);
  layernorm_kernel<<<BB, 256, 0, stream>>>(t1, ln_ca_g, ln_ca_b, ds1, ds2, EE);

  // ---- GLU MLP ----
  layernorm_kernel<<<BB, 256, 0, stream>>>(ds2, glu_ln0_g, glu_ln0_b, nullptr, z, EE);
  launch_gemm(z, glu_fc0, wg, part, BB, GG, EE, 1.0f, /*gelu*/1, 8, stream);
  launch_gemm(z, glu_fc1, vg, part, BB, GG, EE, 1.0f, 0, 8, stream);
  mul_kernel<<<(int)((BG + 255) / 256), 256, 0, stream>>>(wg, vg, hpre, (int)BG);
  layernorm_kernel<<<BB, 256, 0, stream>>>(hpre, glu_ln1_g, glu_ln1_b, nullptr, hbuf, GG);
  launch_gemm(hbuf, glu_fc2, t2, part, BB, EE, GG, 1.0f, 0, 16, stream);
  add_kernel<<<(int)((BE + 255) / 256), 256, 0, stream>>>(ds2, t2, out_dec, (int)BE);
}
